// SpikeRefractoryEncoder_68702296867228
// MI455X (gfx1250) — compile-verified, matrix operand order checked by reading, not run on hardware
//
#include <hip/hip_runtime.h>
#include <math.h>

// ---------------------------------------------------------------------------
// SpikeRefractoryEncoder for MI455X (gfx1250, wave32)
//   Kernel 1: per-observation MLP -> spike_raw, via v_wmma_f32_16x16x32_f16,
//             fragments assembled with b128 LDS loads (transposed weights).
//   Kernel 2: O(N^2) refractory decay sum; LDS staged via
//             GLOBAL_LOAD_ASYNC_TO_LDS_B32 (ASYNCcnt) when available.
// ---------------------------------------------------------------------------

typedef __attribute__((ext_vector_type(16))) _Float16 v16h;
typedef __attribute__((ext_vector_type(8)))  _Float16 v8h;
typedef __attribute__((ext_vector_type(8)))  float    v8f;

#define BDIM   8
#define NDIM   2048
#define NVARS  64
#define HDIM   32
#define DIN    35      // 3 + H
#define KPAD   64      // DIN padded to 2 WMMA K-chunks of 32

// 16-bit A/B fragment for 16x16x32 WMMA: lane = M (A) / N (B), hi = lane>>4.
// Element e maps to k = (e>>3)*16 + hi*8 + (e&7): two contiguous 8-half
// (16-byte) groups per lane -> two aligned b128 LDS loads.
__device__ __forceinline__ v16h load_frag(const _Float16* rowp, int hi) {
    const v8h lo = *(const v8h*)(rowp + hi * 8);
    const v8h hh = *(const v8h*)(rowp + 16 + hi * 8);
    return __builtin_shufflevector(lo, hh, 0, 1, 2, 3, 4, 5, 6, 7,
                                           8, 9, 10, 11, 12, 13, 14, 15);
}

__global__ __launch_bounds__(256)
void spike_mlp_wmma(const float* __restrict__ value,
                    const float* __restrict__ timen,
                    const float* __restrict__ mask,
                    const int*   __restrict__ var_id,
                    const float* __restrict__ var_emb,
                    const float* __restrict__ W1,
                    const float* __restrict__ b1,
                    const float* __restrict__ W2,
                    const float* __restrict__ b2,
                    const float* __restrict__ gate_w,
                    const float* __restrict__ gate_b,
                    float* __restrict__ spike_raw)
{
    // Transposed weights: [col][k] so a lane's B-fragment is contiguous in k.
    __shared__ __align__(16) _Float16 sW1T[HDIM][KPAD];   // 4KB (k>=DIN zeroed)
    __shared__ __align__(16) _Float16 sW2T[HDIM][HDIM];   // 2KB
    __shared__ __align__(16) _Float16 sF  [8][16][KPAD];  // 16KB feats / wave tile
    __shared__ __align__(16) _Float16 sH1 [8][16][HDIM];  // 8KB gelu(h1)
    __shared__ float                  sH2 [8][16][HDIM];  // 16KB h2

    const int tid  = threadIdx.x;
    const int wave = tid >> 5;
    const int lane = tid & 31;
    const int hi   = lane >> 4;
    const int l15  = lane & 15;

    // ---- stage weights (transposed, f16) ----
    for (int i = tid; i < KPAD * HDIM; i += 256) {
        const int k = i >> 5, c = i & 31;
        sW1T[c][k] = (_Float16)((k < DIN) ? W1[k * HDIM + c] : 0.0f);
    }
    for (int i = tid; i < HDIM * HDIM; i += 256) {
        const int k = i >> 5, c = i & 31;
        sW2T[c][k] = (_Float16)W2[k * HDIM + c];
    }

    // ---- stage features: lane handles row=l15, K half = hi*32..hi*32+31 ----
    const int rowBase = blockIdx.x * 128 + wave * 16;   // grid covers B*N exactly
    {
        const int   r   = l15;
        const int   g   = rowBase + r;
        const float val = value[g];
        const float tt  = timen[g];
        const float mk  = mask[g];
        const int   vid = var_id[g];
        const int   k0  = hi * 32;
        for (int kk = 0; kk < 32; ++kk) {
            const int k = k0 + kk;
            float f;
            if      (k == 0)   f = val * mk;
            else if (k == 1)   f = tt;
            else if (k == 2)   f = mk;
            else if (k < DIN)  f = var_emb[vid * HDIM + (k - 3)];
            else               f = 0.0f;
            sF[wave][r][k] = (_Float16)f;
        }
    }
    __syncthreads();

    // ---- GEMM1: h1 = gelu(feats @ W1 + b1); 2 col-tiles x 2 K-chunks ----
    const v16h a0 = load_frag(&sF[wave][l15][0],  hi);
    const v16h a1 = load_frag(&sF[wave][l15][32], hi);
    for (int t = 0; t < 2; ++t) {
        const int  col = l15 + t * 16;
        const v16h bw0 = load_frag(&sW1T[col][0],  hi);
        const v16h bw1 = load_frag(&sW1T[col][32], hi);
        v8f c;
        const float bias = b1[col];
        for (int v = 0; v < 8; ++v) c[v] = bias;
        c = __builtin_amdgcn_wmma_f32_16x16x32_f16(false, a0, false, bw0, (short)0, c, false, false);
        c = __builtin_amdgcn_wmma_f32_16x16x32_f16(false, a1, false, bw1, (short)0, c, false, false);
        // exact GELU; C layout: VGPR v -> row v + hi*8, N = col
        for (int v = 0; v < 8; ++v) {
            const float x  = c[v];
            const float gl = 0.5f * x * (1.0f + erff(x * 0.70710678118654752f));
            sH1[wave][v + hi * 8][col] = (_Float16)gl;
        }
    }
    __syncthreads();

    // ---- GEMM2: h2 = h1 @ W2 + b2 (K = 32, single chunk) ----
    const v16h a2 = load_frag(&sH1[wave][l15][0], hi);
    for (int t = 0; t < 2; ++t) {
        const int  col = l15 + t * 16;
        const v16h bw  = load_frag(&sW2T[col][0], hi);
        v8f c;
        const float bias = b2[col];
        for (int v = 0; v < 8; ++v) c[v] = bias;
        c = __builtin_amdgcn_wmma_f32_16x16x32_f16(false, a2, false, bw, (short)0, c, false, false);
        for (int v = 0; v < 8; ++v) sH2[wave][v + hi * 8][col] = c[v];
    }
    __syncthreads();

    // ---- gate: spike_raw = sigmoid(h2 . gate_w + gate_b) * mask ----
    {
        const int r = l15;
        float partial = 0.0f;
        for (int j = 0; j < 16; ++j) {
            const int cidx = hi * 16 + j;
            partial += sH2[wave][r][cidx] * gate_w[cidx];
        }
        const float other = __shfl_xor(partial, 16, 32);  // swap lane halves
        if (hi == 0) {
            const float z = partial + other + gate_b[0];
            const int   g = rowBase + r;
            const float s = 1.0f / (1.0f + __expf(-z));
            spike_raw[g] = s * mask[g];
        }
    }
}

// ---------------------------------------------------------------------------
// Async global->LDS staging (gfx1250 GLOBAL_LOAD_ASYNC_TO_LDS_B32, ASYNCcnt).
// Builtin params are int* in AS1 (global) / AS3 (LDS) per the clang diagnostic.
// Guarded so compilation never regresses if the builtin is absent.
// ---------------------------------------------------------------------------
#if defined(__gfx1250__) && __has_builtin(__builtin_amdgcn_global_load_async_to_lds_b32)
#define HAVE_ASYNC_LDS 1
#define ASYNC_B32(gp, lp)                                                    \
    __builtin_amdgcn_global_load_async_to_lds_b32(                           \
        (__attribute__((address_space(1))) int*)(gp),                        \
        (__attribute__((address_space(3))) int*)(lp), 0, 0)
#else
#define HAVE_ASYNC_LDS 0
#endif

__device__ __forceinline__ void wait_async_zero() {
#if HAVE_ASYNC_LDS
#if __has_builtin(__builtin_amdgcn_s_wait_asynccnt)
    __builtin_amdgcn_s_wait_asynccnt(0);
#else
    asm volatile("s_wait_asynccnt 0" ::: "memory");
#endif
#endif
}

// ---------------------------------------------------------------------------
// Kernel 2: refractory[b,n] = sum_m [var==, t_m<t_n] exp((t_m-t_n)/tau_n)*sr[m]
// out = sr[n] * (1 - tanh(alpha_n * refractory)) * mask[n]
// One block = 256 consecutive n's of one batch row; full row staged in LDS.
// (mask_m is already folded into spike_raw; mask_n is absorbed by the final
//  *mask, so the pair predicate is just strict-earlier + same-variable.)
// ---------------------------------------------------------------------------
__global__ __launch_bounds__(256)
void spike_refractory(const float* __restrict__ timen,
                      const float* __restrict__ mask,
                      const int*   __restrict__ var_id,
                      const float* __restrict__ log_tau_r,
                      const float* __restrict__ log_alpha,
                      const float* __restrict__ spike_raw,
                      float* __restrict__ out)
{
    __shared__ float st[NDIM];
    __shared__ float ss[NDIM];
    __shared__ int   sv[NDIM];

    const int tid   = threadIdx.x;
    const int b     = blockIdx.x >> 3;   // NDIM/256 = 8 chunks per batch row
    const int chunk = blockIdx.x & 7;
    const int base  = b * NDIM;

    for (int i = tid; i < NDIM; i += 256) {
#if HAVE_ASYNC_LDS
        ASYNC_B32(timen     + base + i, &st[i]);
        ASYNC_B32(spike_raw + base + i, &ss[i]);
        ASYNC_B32(var_id    + base + i, &sv[i]);
#else
        st[i] = timen[base + i];
        ss[i] = spike_raw[base + i];
        sv[i] = var_id[base + i];
#endif
    }
    wait_async_zero();
    __syncthreads();

    const int   n       = chunk * 256 + tid;
    const float tn      = st[n];
    const int   vn      = sv[n];
    const float tau     = fmaxf(__expf(log_tau_r[vn]), 1e-6f);
    const float inv_tau = 1.0f / tau;
    const float alpha   = __expf(log_alpha[vn]);

    float acc = 0.0f;
    for (int m = 0; m < NDIM; ++m) {
        const float tm = st[m];
        const bool  ok = (tm < tn) & (sv[m] == vn);   // strict earlier, same var
        const float w  = ok ? __expf((tm - tn) * inv_tau) : 0.0f;
        acc += w * ss[m];
    }

    const float inhibit = tanhf(alpha * acc);
    out[base + n] = ss[n] * (1.0f - inhibit) * mask[base + n];
}

// ---------------------------------------------------------------------------
extern "C" void kernel_launch(void* const* d_in, const int* in_sizes, int n_in,
                              void* d_out, int out_size, void* d_ws, size_t ws_size,
                              hipStream_t stream)
{
    const float* value     = (const float*)d_in[0];
    const float* time_norm = (const float*)d_in[1];
    const float* mask      = (const float*)d_in[2];
    const int*   var_id    = (const int*)  d_in[3];
    const float* var_emb   = (const float*)d_in[4];
    const float* W1        = (const float*)d_in[5];
    const float* b1        = (const float*)d_in[6];
    const float* W2        = (const float*)d_in[7];
    const float* b2        = (const float*)d_in[8];
    const float* gate_w    = (const float*)d_in[9];
    const float* gate_b    = (const float*)d_in[10];
    const float* log_tau_r = (const float*)d_in[11];
    const float* log_alpha = (const float*)d_in[12];

    float* out = (float*)d_out;
    float* sr  = (float*)d_ws;          // B*N floats = 64KB of workspace

    // Kernel 1: 128 rows per block (8 waves x 16-row WMMA tiles)
    spike_mlp_wmma<<<(BDIM * NDIM) / 128, 256, 0, stream>>>(
        value, time_norm, mask, var_id, var_emb,
        W1, b1, W2, b2, gate_w, gate_b, sr);

    // Kernel 2: one batch row per 8 blocks
    spike_refractory<<<BDIM * (NDIM / 256), 256, 0, stream>>>(
        time_norm, mask, var_id, log_tau_r, log_alpha, sr, out);
}
